// RNN2_23441931501744
// MI455X (gfx1250) — matrix operand order repaired and checked by reading, not compiled, not run on hardware
//
#include <hip/hip_runtime.h>
#include <hip/hip_bf16.h>

// ---------------------------------------------------------------------------
// Problem constants (match reference)
// ---------------------------------------------------------------------------
#define TSTEPS   512
#define BATCH    64
#define XF       160      // input_dim * num_stations = 8*20
#define LATENT   512
#define OUTD     8
#define NSTA     20
#define KT_X     5               // 160 / 32
#define KT_H     16              // 512 / 32
#define KT_ALL   21

typedef __attribute__((ext_vector_type(16))) __bf16 v16bf;
typedef __attribute__((ext_vector_type(8)))  float  v8f;

union FragU { uint4 u[2]; v16bf v; };
union H8    { uint4 u;    __bf16 b[8]; };

// d_ws layout (bytes):
//   [0,          688128)  wiB : Wi packed bf16 B-fragments (672*512)
//   [688128,     704512)  WoT : Wo transposed fp32 [8][512]
//   [704512,   11190272)  xA  : x packed bf16 A-fragments (T*5*4*32*16) [optional]
#define WS_WIB_ELEMS  (KT_ALL * 32 * 32 * 16)          // 344064 bf16
#define WS_WOT_OFF    (WS_WIB_ELEMS * 2)               // 688128 B
#define WS_XA_OFF     (WS_WOT_OFF + LATENT * OUTD * 4) // 704512 B
#define WS_XA_ELEMS   (TSTEPS * KT_X * 4 * 32 * 16)    // 5,242,880 bf16
#define WS_NEED_PX    (WS_XA_OFF + (size_t)WS_XA_ELEMS * 2)

// ---------------------------------------------------------------------------
// Pack Wi (672x512 f32 row-major) -> bf16 B-fragment order.
// 16-bit B 32x16 layout: lane<16: N=nt*16+lane, K=kt*32+{0..7,16..23};
//                        lane>=16: N=nt*16+lane-16, K=kt*32+{8..15,24..31}.
// ---------------------------------------------------------------------------
__global__ void rnn_pack_wi(const float* __restrict__ Wi,
                            __hip_bfloat16* __restrict__ wiB_raw) {
    __bf16* wiB = reinterpret_cast<__bf16*>(wiB_raw);
    int f = blockIdx.x * 256 + threadIdx.x;            // 0 .. 344063
    if (f >= WS_WIB_ELEMS) return;
    int e    = f & 15;
    int lane = (f >> 4) & 31;
    int nt   = (f >> 9) & 31;
    int kt   = f >> 14;
    int n = nt * 16 + (lane & 15);
    int k = kt * 32 + ((lane >> 4) ? 8 : 0) + (e < 8 ? e : e + 8);
    wiB[f] = (__bf16)Wi[k * LATENT + n];
}

// Transpose Wo [512][8] -> WoT [8][512] (fp32) for contiguous decode loads.
__global__ void rnn_pack_wo(const float* __restrict__ Wo,
                            float* __restrict__ WoT) {
    int f = blockIdx.x * 256 + threadIdx.x;            // 0 .. 4095
    if (f >= LATENT * OUTD) return;
    int o = f >> 9;          // WoT[o][j]
    int j = f & 511;
    WoT[f] = Wo[j * OUTD + o];
}

// Pack x [T][64][160] f32 -> bf16 A-fragment order.
// 16-bit A 16x32 layout: lane row m = mt*16 + lane%16; per-lane 16 elems:
//   e<8  -> K = kt*32 + (lane>=16 ? 8:0) + e        (VGPR0..3)
//   e>=8 -> K = kt*32 + (lane>=16 ? 8:0) + 16 + e-8 (VGPR4..7)
__global__ void rnn_pack_x(const float* __restrict__ x,
                           __hip_bfloat16* __restrict__ xA_raw) {
    __bf16* xA = reinterpret_cast<__bf16*>(xA_raw);
    int f = blockIdx.x * 256 + threadIdx.x;            // 0 .. 5,242,879
    if (f >= WS_XA_ELEMS) return;
    int e    = f & 15;
    int lane = (f >> 4) & 31;
    int mt   = (f >> 9) & 3;
    int tk   = f >> 11;
    int kt   = tk % KT_X;
    int t    = tk / KT_X;
    int m = mt * 16 + (lane & 15);
    int k = kt * 32 + ((lane >> 4) ? 8 : 0) + (e < 8 ? e : e + 8);
    xA[f] = (__bf16)x[((size_t)t * BATCH + m) * XF + k];
}

// ---------------------------------------------------------------------------
// Deferred decode, phase 1: all 1024 threads, half-dot partials into psum.
// hbuf holds h_t in XOR-swizzled bf16 chunks (chunk' = chunk ^ (row&15)).
// ---------------------------------------------------------------------------
__device__ __forceinline__ void decode_partial(const __bf16* hbuf, float* psum,
                                               const float* __restrict__ WoT,
                                               int tid) {
    const int mo = tid & 511;
    const int p  = tid >> 9;              // which half of the dot
    const int m  = mo >> 3;
    const int o  = mo & 7;
    const int swm = m & 15;
    const float*  wt   = WoT + (o << 9) + (p << 8);
    const __bf16* hrow = hbuf + (m << 9);
    float s0 = 0.f, s1 = 0.f;
#pragma unroll 4
    for (int jj = 0; jj < 256; jj += 8) {
        const int j = (p << 8) + jj;
        H8 hh; hh.u = *(const uint4*)(&hrow[((j >> 3) ^ swm) << 3]);
        float4 w0 = *(const float4*)(wt + jj);
        float4 w1 = *(const float4*)(wt + jj + 4);
        s0 += (float)hh.b[0]*w0.x + (float)hh.b[1]*w0.y +
              (float)hh.b[2]*w0.z + (float)hh.b[3]*w0.w;
        s1 += (float)hh.b[4]*w1.x + (float)hh.b[5]*w1.y +
              (float)hh.b[6]*w1.z + (float)hh.b[7]*w1.w;
    }
    psum[tid] = s0 + s1;
}

// Phase 2 (after barrier): threads < 512 combine halves, sigmoid, broadcast.
__device__ __forceinline__ void decode_store(const float* psum,
                                             const float* __restrict__ bo,
                                             float* __restrict__ out,
                                             int tdec, int tid) {
    if (tid < BATCH * OUTD) {
        const int m = tid >> 3;
        const int o = tid & 7;
        const float s  = psum[tid] + psum[tid + 512] + bo[o];
        const float ov = 1.0f / (1.0f + __expf(-s));
        float* op = out + (((size_t)tdec * BATCH + m) * NSTA) * OUTD + o;
#pragma unroll
        for (int sta = 0; sta < NSTA; ++sta) op[sta * OUTD] = ov;
    }
}

// ---------------------------------------------------------------------------
// Persistent single-WGP recurrent scan. 32 waves; wave w owns N columns
// [16w,16w+16). PX: x pre-packed as bf16 A-fragments (no cvt in hot loop).
// ---------------------------------------------------------------------------
template <bool PX>
__global__ __launch_bounds__(1024, 1)
void rnn_persistent(const float* __restrict__ x,              // [T,64,160]
                    const __hip_bfloat16* __restrict__ xA_raw,
                    const float* __restrict__ h0,             // [64,512]
                    const __hip_bfloat16* __restrict__ wiB_raw,
                    const float* __restrict__ bi,             // [512]
                    const float* __restrict__ WoT,            // [8,512]
                    const float* __restrict__ bo,             // [8]
                    float* __restrict__ out) {                // [T,64,20,8]
    __shared__ __align__(16) __bf16 hbuf[BATCH * LATENT];     // 64 KB
    __shared__ float psum[1024];                              // 4 KB

    const __bf16* wiB = reinterpret_cast<const __bf16*>(wiB_raw);
    const __bf16* xA  = reinterpret_cast<const __bf16*>(xA_raw);
    const int tid = threadIdx.x;
    const int w   = tid >> 5;
    const int l   = tid & 31;
    const int lq  = l & 15;
    const int lh  = l >> 4;

    // ---- initial h: fp32 -> bf16, swizzled ----
    for (int i = tid; i < BATCH * LATENT; i += 1024) {
        int m = i >> 9;
        int c = i & 511;
        int sw = (c >> 3) ^ (m & 15);
        hbuf[(m << 9) + (sw << 3) + (c & 7)] = (__bf16)h0[i];
    }
    __syncthreads();

    const float bi_l = bi[w * 16 + lq];
    const int   col  = w * 16 + lq;

    for (int t = 0; t < TSTEPS; ++t) {
        v8f acc[4] = {};
        const float* xrow0 = x + (size_t)t * BATCH * XF;

        // ---- K-tiles from x_t ----
        for (int kt = 0; kt < KT_X; ++kt) {
            const __bf16* bp = wiB + ((((size_t)kt * 32 + w) * 32 + l) << 4);
            FragU bf;
            bf.u[0] = *(const uint4*)(bp);
            bf.u[1] = *(const uint4*)(bp + 8);
            __builtin_prefetch(bp + 32 * 32 * 16, 0, 0);
#pragma unroll
            for (int mt = 0; mt < 4; ++mt) {
                FragU af;
                if (PX) {
                    const __bf16* ap =
                        xA + ((((size_t)(t * KT_X + kt)) * 4 + mt) * 32 + l) * 16;
                    af.u[0] = *(const uint4*)(ap);
                    af.u[1] = *(const uint4*)(ap + 8);
                } else {
                    const int m  = mt * 16 + lq;
                    const int k0 = kt * 32 + (lh ? 8 : 0);
                    const float* xp = xrow0 + m * XF + k0;
                    float4 f0 = *(const float4*)(xp);
                    float4 f1 = *(const float4*)(xp + 4);
                    float4 f2 = *(const float4*)(xp + 16);
                    float4 f3 = *(const float4*)(xp + 20);
                    v16bf a;
                    a[0]=(__bf16)f0.x;  a[1]=(__bf16)f0.y;  a[2]=(__bf16)f0.z;  a[3]=(__bf16)f0.w;
                    a[4]=(__bf16)f1.x;  a[5]=(__bf16)f1.y;  a[6]=(__bf16)f1.z;  a[7]=(__bf16)f1.w;
                    a[8]=(__bf16)f2.x;  a[9]=(__bf16)f2.y;  a[10]=(__bf16)f2.z; a[11]=(__bf16)f2.w;
                    a[12]=(__bf16)f3.x; a[13]=(__bf16)f3.y; a[14]=(__bf16)f3.z; a[15]=(__bf16)f3.w;
                    af.v = a;
                }
                acc[mt] = __builtin_amdgcn_wmma_f32_16x16x32_bf16(
                    false, af.v, false, bf.v, (short)0, acc[mt], false, false);
            }
        }

        // ---- K-tiles from h_prev (LDS, swizzled ds_load_b128) ----
        for (int kt = 0; kt < KT_H; ++kt) {
            const __bf16* bp =
                wiB + (((((size_t)kt + KT_X) * 32 + w) * 32 + l) << 4);
            FragU bf;
            bf.u[0] = *(const uint4*)(bp);
            bf.u[1] = *(const uint4*)(bp + 8);
            __builtin_prefetch(bp + 32 * 32 * 16, 0, 0);
            const int j0 = kt * 32 + (lh ? 8 : 0);
            const int c0 = j0 >> 3;
            const int c1 = (j0 + 16) >> 3;
#pragma unroll
            for (int mt = 0; mt < 4; ++mt) {
                const int m  = mt * 16 + lq;
                const int sw = m & 15;
                FragU af;
                af.u[0] = *(const uint4*)(&hbuf[(m << 9) + ((c0 ^ sw) << 3)]);
                af.u[1] = *(const uint4*)(&hbuf[(m << 9) + ((c1 ^ sw) << 3)]);
                acc[mt] = __builtin_amdgcn_wmma_f32_16x16x32_bf16(
                    false, af.v, false, bf.v, (short)0, acc[mt], false, false);
            }
        }

        // ---- deferred decode of out[t-1] (reads hbuf = h_{t-1}); overlaps
        //      the WMMA phase and reuses the store barrier below ----
        if (t > 0) decode_partial(hbuf, psum, WoT, tid);

        __syncthreads();   // hbuf reads (GEMM + decode) done; psum visible

        // ---- tanh + store h_t (C layout: elem r -> row mt*16+r+8*lh, col lq)
#pragma unroll
        for (int mt = 0; mt < 4; ++mt) {
#pragma unroll
            for (int r = 0; r < 8; ++r) {
                const int m  = mt * 16 + r + 8 * lh;
                const float hv = tanhf(acc[mt][r] + bi_l);
                const int sw = (col >> 3) ^ (m & 15);
                hbuf[(m << 9) + (sw << 3) + (col & 7)] = (__bf16)hv;
            }
        }
        if (t > 0) decode_store(psum, bo, out, t - 1, tid);

        __syncthreads();   // h_t visible; psum consumed
    }

    // ---- final decode: out[T-1] from hbuf = h_{T-1} ----
    decode_partial(hbuf, psum, WoT, tid);
    __syncthreads();
    decode_store(psum, bo, out, TSTEPS - 1, tid);
}

// ---------------------------------------------------------------------------
// Host launcher
// ---------------------------------------------------------------------------
extern "C" void kernel_launch(void* const* d_in, const int* in_sizes, int n_in,
                              void* d_out, int out_size, void* d_ws, size_t ws_size,
                              hipStream_t stream) {
    // setup_inputs order: x, h, num_stations, Wi, bi, Wo, bo
    const float* x  = (const float*)d_in[0];
    const float* h0 = (const float*)d_in[1];
    const float* Wi = (const float*)d_in[3];
    const float* bi = (const float*)d_in[4];
    const float* Wo = (const float*)d_in[5];
    const float* bo = (const float*)d_in[6];

    char* ws = (char*)d_ws;
    __hip_bfloat16* wiB = (__hip_bfloat16*)ws;
    float*          WoT = (float*)(ws + WS_WOT_OFF);
    __hip_bfloat16* xA  = (__hip_bfloat16*)(ws + WS_XA_OFF);

    rnn_pack_wi<<<(WS_WIB_ELEMS + 255) / 256, 256, 0, stream>>>(Wi, wiB);
    rnn_pack_wo<<<(LATENT * OUTD + 255) / 256, 256, 0, stream>>>(Wo, WoT);

    if (ws_size >= WS_NEED_PX) {
        rnn_pack_x<<<(WS_XA_ELEMS + 255) / 256, 256, 0, stream>>>(x, xA);
        rnn_persistent<true><<<1, 1024, 0, stream>>>(x, xA, h0, wiB, bi, WoT,
                                                     bo, (float*)d_out);
    } else {
        rnn_persistent<false><<<1, 1024, 0, stream>>>(x, xA, h0, wiB, bi, WoT,
                                                      bo, (float*)d_out);
    }
}
